// GPT2Attention_w_Adapter_60507499266300
// MI455X (gfx1250) — matrix operand order repaired and checked by reading, not compiled, not run on hardware
//
#include <hip/hip_runtime.h>
#include <hip/hip_bf16.h>
#include <math.h>

// ---------------------------------------------------------------------------
// Problem constants (GPT2 attention w/ adapter): B=2, S=2048, A=64, D=1024,
// H=16, HD=64.  M_qkv = B*S + B*A = 4224.
// ---------------------------------------------------------------------------
#define BATCH     2
#define SEQ       2048
#define ADA       64
#define DIM       1024
#define HEADS     16
#define HD        64
#define MROWS     (BATCH * SEQ)            // 4096
#define MQKV      (MROWS + BATCH * ADA)    // 4224
#define N3D       (3 * DIM)                // 3072

typedef __bf16 bf16_t;
typedef __attribute__((ext_vector_type(16))) __bf16 v16bf;
typedef __attribute__((ext_vector_type(8)))  float  v8f;
typedef int v4i __attribute__((vector_size(16)));   // matches builtin param type

// CDNA5 async global->LDS copy path (ASYNCcnt), guarded so compilation never
// depends on the builtin being present in this toolchain.
#if defined(__has_builtin)
#if __has_builtin(__builtin_amdgcn_global_load_async_to_lds_b128) && \
    __has_builtin(__builtin_amdgcn_s_wait_asynccnt)
#define HAVE_ASYNC_LDS 1
#endif
#endif
#ifndef HAVE_ASYNC_LDS
#define HAVE_ASYNC_LDS 0
#endif

// ---------------------------------------------------------------------------
// WMMA helper: D = A(16x32 bf16) x B(32x16 bf16) + C(16x16 f32)
// ---------------------------------------------------------------------------
__device__ __forceinline__ v8f wmma_bf16(v16bf a, v16bf b, v8f c) {
  return __builtin_amdgcn_wmma_f32_16x16x32_bf16(
      /*neg_a=*/false, a, /*neg_b=*/false, b,
      /*c_mod=*/(short)0, c, /*reuse_a=*/false, /*reuse_b=*/false);
}

// A-fragment K index for a 16x32 16-bit A matrix (contiguous order):
// lanes 0-15 hold K {0..15}; lanes 16-31 hold K {16..31}.
__device__ __forceinline__ int a_kidx(int hi, int e) {
  return hi * 16 + e;
}

// Convert 16 contiguous f32 to a bf16 vector (B-fragment per-lane image).
__device__ __forceinline__ v16bf cvt16(const float* __restrict__ p) {
  v16bf v;
#pragma unroll
  for (int i = 0; i < 16; ++i) v[i] = (bf16_t)p[i];
  return v;
}

// Convert two 8-float contiguous segments into an A-fragment lane image.
__device__ __forceinline__ v16bf cvtA(const float* __restrict__ s0,
                                      const float* __restrict__ s1) {
  v16bf v;
#pragma unroll
  for (int i = 0; i < 8; ++i) {
    v[i]     = (bf16_t)s0[i];
    v[8 + i] = (bf16_t)s1[i];
  }
  return v;
}

// ---------------------------------------------------------------------------
// Pack f32 row-major [M x K] (ld = row stride) into bf16 A-fragment images:
//   dst[((mt*Kt + kt)*32 + lane)*16 + e] = src[(mt*16 + lane&15)*ld + kt*32 + aK]
// ---------------------------------------------------------------------------
__global__ __launch_bounds__(256) void pack_a_kernel(
    bf16_t* __restrict__ dst, const float* __restrict__ src,
    int ld, int Kt, int total) {
  int idx = blockIdx.x * 256 + threadIdx.x;
  if (idx >= total) return;
  int e    = idx & 15;
  int lane = (idx >> 4) & 31;
  int t    = idx >> 9;
  int kt   = t % Kt;
  int mt   = t / Kt;
  int m    = mt * 16 + (lane & 15);
  int hi   = (lane >> 4) & 1;
  int k    = kt * 32 + a_kidx(hi, e);
  dst[idx] = (bf16_t)src[(size_t)m * ld + k];
}

// ---------------------------------------------------------------------------
// Pack f32 row-major [K x N] into bf16 B-fragment images:
//   lane&15 -> n-in-tile, lane>>4 selects K half, e -> K within half.
// ---------------------------------------------------------------------------
__global__ __launch_bounds__(256) void pack_b_kernel(
    bf16_t* __restrict__ dst, const float* __restrict__ src,
    int ld, int Kt, int total) {
  int idx = blockIdx.x * 256 + threadIdx.x;
  if (idx >= total) return;
  int e    = idx & 15;
  int lane = (idx >> 4) & 31;
  int t    = idx >> 9;
  int kt   = t % Kt;
  int nt   = t / Kt;
  int n    = nt * 16 + (lane & 15);
  int k    = kt * 32 + ((lane >> 4) & 1) * 16 + e;
  dst[idx] = (bf16_t)src[(size_t)k * ld + n];
}

// ---------------------------------------------------------------------------
// Pack V (or adapter V) from the qkv buffer into per-(b,h) B-fragment images.
// V as a GEMM B operand:  K-dim = key index, N-dim = head-dim (64 -> 4 tiles).
//   dst[bh][nt][kt][lane][e];  src = qkv[(rowOff + b*rowPerB + k)*3072
//                                        + colbase + h*64 + n]
// ---------------------------------------------------------------------------
__global__ __launch_bounds__(256) void pack_v_kernel(
    bf16_t* __restrict__ dst, const float* __restrict__ qkv,
    int rowPerB, int rowOff, int Kt, int perBH, int colbase) {
  int idx = blockIdx.x * 256 + threadIdx.x;
  if (idx >= perBH) return;
  int bh   = blockIdx.y;
  int b    = bh >> 4;
  int h    = bh & 15;
  int e    = idx & 15;
  int lane = (idx >> 4) & 31;
  int t    = idx >> 9;
  int kt   = t % Kt;
  int n    = ((t / Kt) * 16) + (lane & 15);
  int k    = kt * 32 + ((lane >> 4) & 1) * 16 + e;
  dst[(size_t)bh * perBH + idx] =
      (bf16_t)qkv[(size_t)(rowOff + b * rowPerB + k) * N3D + colbase + h * 64 + n];
}

// ---------------------------------------------------------------------------
// bf16 WMMA GEMM:  C[M x N] = A[M x K] * B[K x N] + bias[N]   (f32 out)
// A, B pre-packed as fragment images.  Wave tile: 16x64 (4 wmma accumulators);
// block = 8 waves (4 along M x 2 along N) -> 64 x 128 block tile.
// B tiles (8 KB per 32-deep K chunk, shared by all 8 waves) are staged through
// LDS with a double buffer; stage k+1 runs asynchronously
// (GLOBAL_LOAD_ASYNC_TO_LDS_B128 / ASYNCcnt) while WMMAs consume chunk k.
// ---------------------------------------------------------------------------
__device__ __forceinline__ void stage_b_tiles(
    const bf16_t* __restrict__ Bp, bf16_t* __restrict__ Bs_buf,
    int ntb, int Kt, int kt) {
  // 8 tiles x 1 KB = 512 units of 16 B; 2 units per thread (256 threads).
  const int u0 = threadIdx.x * 2;
#pragma unroll
  for (int i = 0; i < 2; ++i) {
    int u = u0 + i;
    int j = u >> 6;                       // tile index 0..7 (64 units/tile)
    size_t srcoff = (((size_t)(ntb + j) * Kt + kt) * 64 + (u & 63)) * 16;
    const char* src = (const char*)Bp + srcoff;
    char* dst = (char*)Bs_buf + (size_t)u * 16;
#if HAVE_ASYNC_LDS
    __builtin_amdgcn_global_load_async_to_lds_b128(
        (__attribute__((address_space(1))) v4i*)src,
        (__attribute__((address_space(3))) v4i*)dst, 0, 0);
#else
    *(int4*)dst = *(const int4*)src;
#endif
  }
}

__global__ __launch_bounds__(256) void gemm_bf16_kernel(
    const bf16_t* __restrict__ Ap, const bf16_t* __restrict__ Bp,
    const float* __restrict__ bias, float* __restrict__ C,
    int M, int N, int K) {
  __shared__ __align__(16) bf16_t Bs[2][8][32][16];   // 16 KB double buffer

  const int lane  = threadIdx.x & 31;
  const int wave  = threadIdx.x >> 5;
  const int wm    = wave & 3;           // 4 waves along M
  const int wn    = wave >> 2;          // 2 waves along N
  const int mtile = blockIdx.x * 4 + wm;
  const int ntb   = blockIdx.y * 8;     // block's first 16-wide N tile
  const int Kt    = K >> 5;

  const v16bf* Abase = (const v16bf*)Ap + ((size_t)mtile * Kt) * 32 + lane;

  v8f acc0 = {}, acc1 = {}, acc2 = {}, acc3 = {};

  stage_b_tiles(Bp, &Bs[0][0][0][0], ntb, Kt, 0);
  for (int kt = 0; kt < Kt; ++kt) {
#if HAVE_ASYNC_LDS
    __builtin_amdgcn_s_wait_asynccnt(0);
#endif
    __syncthreads();                    // buffer (kt&1) is ready for everyone
    if (kt + 1 < Kt)
      stage_b_tiles(Bp, &Bs[(kt + 1) & 1][0][0][0], ntb, Kt, kt + 1);

    v16bf a = Abase[(size_t)kt * 32];
    if (kt + 4 < Kt)
      __builtin_prefetch((const void*)(Abase + (size_t)(kt + 4) * 32), 0, 3);

    const bf16_t* bs = &Bs[kt & 1][wn * 4][0][0];
    v16bf b0 = *(const v16bf*)(bs + ((size_t)(0 * 32 + lane)) * 16);
    v16bf b1 = *(const v16bf*)(bs + ((size_t)(1 * 32 + lane)) * 16);
    v16bf b2 = *(const v16bf*)(bs + ((size_t)(2 * 32 + lane)) * 16);
    v16bf b3 = *(const v16bf*)(bs + ((size_t)(3 * 32 + lane)) * 16);

    acc0 = wmma_bf16(a, b0, acc0);
    acc1 = wmma_bf16(a, b1, acc1);
    acc2 = wmma_bf16(a, b2, acc2);
    acc3 = wmma_bf16(a, b3, acc3);
  }

  // C/D layout: lane&15 -> n-in-tile, (lane>>4)*8 + r -> m-in-tile.
  const int hi   = (lane >> 4) & 1;
  const int n_in = lane & 15;
  const int nt0  = ntb + wn * 4;
#pragma unroll
  for (int r = 0; r < 8; ++r) {
    int m = mtile * 16 + hi * 8 + r;
    float* row = C + (size_t)m * N;
    int nb = nt0 * 16 + n_in;
    row[nb]      = acc0[r] + bias[nb];
    row[nb + 16] = acc1[r] + bias[nb + 16];
    row[nb + 32] = acc2[r] + bias[nb + 32];
    row[nb + 48] = acc3[r] + bias[nb + 48];
  }
}

// ---------------------------------------------------------------------------
// Flash attention + adapter attention + tanh gate.
// One wave handles a 16-query tile for one (b,h).  Block = 8 waves.
//   grid = (SEQ/16/8, B*H)
// ---------------------------------------------------------------------------
__global__ __launch_bounds__(256) void attn_kernel(
    const float* __restrict__ qkv, const bf16_t* __restrict__ Vp,
    const bf16_t* __restrict__ Vap, const float* __restrict__ gate,
    float* __restrict__ attnout) {
  __shared__ float Pl[8][16][32];   // wave-private probability staging tiles

  const int lane = threadIdx.x & 31;
  const int w    = threadIdx.x >> 5;
  const int qb   = blockIdx.x * 8 + w;     // 16-row query tile, 0..127
  const int bh   = blockIdx.y;
  const int b    = bh >> 4;
  const int h    = bh & 15;
  const int m_lo = lane & 15;
  const int hi   = (lane >> 4) & 1;
  const float scale = 0.125f;              // 1/sqrt(64)
  const int kcol = DIM + h * 64;           // k columns in qkv

  // ---- Q A-fragments (16x64 -> two 16x32), f32 -> bf16 in-register --------
  const float* qbase =
      qkv + ((size_t)(b * SEQ + qb * 16 + m_lo)) * N3D + h * 64;
  v16bf qa0 = cvtA(qbase + 0 + hi * 16, qbase + 8 + hi * 16);
  v16bf qa1 = cvtA(qbase + 32 + hi * 16, qbase + 40 + hi * 16);

  // ---- main causal attention ----------------------------------------------
  v8f o0 = {}, o1 = {}, o2 = {}, o3 = {};
  float mrow[8], lrow[8];
#pragma unroll
  for (int r = 0; r < 8; ++r) { mrow[r] = -3.0e38f; lrow[r] = 0.f; }

  const int nch = (qb >> 1) + 1;           // 32-key chunks under causal mask
  for (int kk = 0; kk < nch; ++kk) {
    // ---- scores: Q(16x64) x K^T(64x32)  -> two 16x16 accumulators
    v8f s0 = {}, s1 = {};
    {
      const float* k0 =
          qkv + ((size_t)(b * SEQ + kk * 32 + m_lo)) * N3D + kcol + hi * 16;
      const float* k1 =
          qkv + ((size_t)(b * SEQ + kk * 32 + 16 + m_lo)) * N3D + kcol + hi * 16;
      s0 = wmma_bf16(qa0, cvt16(k0), s0);
      s0 = wmma_bf16(qa1, cvt16(k0 + 32), s0);
      s1 = wmma_bf16(qa0, cvt16(k1), s1);
      s1 = wmma_bf16(qa1, cvt16(k1 + 32), s1);
    }

    // ---- scale + causal mask + online softmax (row reductions over 16 lanes)
    const int mg = qb * 16 + hi * 8;       // global q row for r
    const int n0 = kk * 32 + m_lo;         // global key for s0 column
    float alpha[8];
#pragma unroll
    for (int r = 0; r < 8; ++r) {
      float a0 = s0[r] * scale; if (n0 > mg + r)      a0 = -10000.0f;
      float a1 = s1[r] * scale; if (n0 + 16 > mg + r) a1 = -10000.0f;
      float mx = fmaxf(a0, a1);
#pragma unroll
      for (int off = 1; off < 16; off <<= 1)
        mx = fmaxf(mx, __shfl_xor(mx, off, 32));
      float nm = fmaxf(mrow[r], mx);
      float al = __expf(mrow[r] - nm);
      float p0 = __expf(a0 - nm);
      float p1 = __expf(a1 - nm);
      float rs = p0 + p1;
#pragma unroll
      for (int off = 1; off < 16; off <<= 1) rs += __shfl_xor(rs, off, 32);
      mrow[r]  = nm;
      lrow[r]  = lrow[r] * al + rs;
      alpha[r] = al;
      Pl[w][hi * 8 + r][m_lo]      = p0;   // C-layout -> LDS
      Pl[w][hi * 8 + r][m_lo + 16] = p1;
    }
#pragma unroll
    for (int r = 0; r < 8; ++r) {
      o0[r] *= alpha[r]; o1[r] *= alpha[r]; o2[r] *= alpha[r]; o3[r] *= alpha[r];
    }
    // Cross-lane LDS handoff within the wave: wait for DS stores.
    asm volatile("s_wait_dscnt 0x0" ::: "memory");

    // ---- P (16x32) as A-fragment
    v16bf pa;
#pragma unroll
    for (int e = 0; e < 16; ++e)
      pa[e] = (bf16_t)Pl[w][m_lo][a_kidx(hi, e)];

    // ---- O += P x V(32x64), V pre-packed B-fragments
    const v16bf* vb = (const v16bf*)Vp + ((size_t)bh * 4 * 64 + kk) * 32 + lane;
    o0 = wmma_bf16(pa, vb[0 * 64 * 32], o0);
    o1 = wmma_bf16(pa, vb[1 * 64 * 32], o1);
    o2 = wmma_bf16(pa, vb[2 * 64 * 32], o2);
    o3 = wmma_bf16(pa, vb[3 * 64 * 32], o3);
  }

  // ---- adapter attention (64 keys, no mask) -------------------------------
  v8f ao0 = {}, ao1 = {}, ao2 = {}, ao3 = {};
  float mra[8], lra[8];
#pragma unroll
  for (int r = 0; r < 8; ++r) { mra[r] = -3.0e38f; lra[r] = 0.f; }

  for (int kk = 0; kk < 2; ++kk) {
    v8f s0 = {}, s1 = {};
    {
      const float* k0 =
          qkv + ((size_t)(MROWS + b * ADA + kk * 32 + m_lo)) * N3D + kcol + hi * 16;
      const float* k1 = k0 + (size_t)16 * N3D;
      s0 = wmma_bf16(qa0, cvt16(k0), s0);
      s0 = wmma_bf16(qa1, cvt16(k0 + 32), s0);
      s1 = wmma_bf16(qa0, cvt16(k1), s1);
      s1 = wmma_bf16(qa1, cvt16(k1 + 32), s1);
    }
    float alpha[8];
#pragma unroll
    for (int r = 0; r < 8; ++r) {
      float a0 = s0[r] * scale;
      float a1 = s1[r] * scale;
      float mx = fmaxf(a0, a1);
#pragma unroll
      for (int off = 1; off < 16; off <<= 1)
        mx = fmaxf(mx, __shfl_xor(mx, off, 32));
      float nm = fmaxf(mra[r], mx);
      float al = __expf(mra[r] - nm);
      float p0 = __expf(a0 - nm);
      float p1 = __expf(a1 - nm);
      float rs = p0 + p1;
#pragma unroll
      for (int off = 1; off < 16; off <<= 1) rs += __shfl_xor(rs, off, 32);
      mra[r]   = nm;
      lra[r]   = lra[r] * al + rs;
      alpha[r] = al;
      Pl[w][hi * 8 + r][m_lo]      = p0;
      Pl[w][hi * 8 + r][m_lo + 16] = p1;
    }
#pragma unroll
    for (int r = 0; r < 8; ++r) {
      ao0[r] *= alpha[r]; ao1[r] *= alpha[r]; ao2[r] *= alpha[r]; ao3[r] *= alpha[r];
    }
    asm volatile("s_wait_dscnt 0x0" ::: "memory");
    v16bf pa;
#pragma unroll
    for (int e = 0; e < 16; ++e)
      pa[e] = (bf16_t)Pl[w][m_lo][a_kidx(hi, e)];
    const v16bf* vb = (const v16bf*)Vap + ((size_t)bh * 4 * 2 + kk) * 32 + lane;
    ao0 = wmma_bf16(pa, vb[0 * 2 * 32], ao0);
    ao1 = wmma_bf16(pa, vb[1 * 2 * 32], ao1);
    ao2 = wmma_bf16(pa, vb[2 * 2 * 32], ao2);
    ao3 = wmma_bf16(pa, vb[3 * 2 * 32], ao3);
  }

  // ---- normalize, gate, merge heads, store [B*S, D] f32 -------------------
  const float g = tanhf(gate[h]);
#pragma unroll
  for (int r = 0; r < 8; ++r) {
    float inv  = 1.0f / lrow[r];
    float inva = 1.0f / lra[r];
    size_t row =
        (size_t)(b * SEQ + qb * 16 + hi * 8 + r) * DIM + h * 64 + m_lo;
    attnout[row + 0]  = o0[r] * inv + g * ao0[r] * inva;
    attnout[row + 16] = o1[r] * inv + g * ao1[r] * inva;
    attnout[row + 32] = o2[r] * inv + g * ao2[r] * inva;
    attnout[row + 48] = o3[r] * inv + g * ao3[r] * inva;
  }
}

// ---------------------------------------------------------------------------
// Host-side launch orchestration.
// ---------------------------------------------------------------------------
extern "C" void kernel_launch(void* const* d_in, const int* in_sizes, int n_in,
                              void* d_out, int out_size, void* d_ws, size_t ws_size,
                              hipStream_t stream) {
  (void)in_sizes; (void)n_in; (void)out_size; (void)ws_size;

  const float* hidden   = (const float*)d_in[0];  // [2,2048,1024]
  const float* adapter  = (const float*)d_in[1];  // [2,64,1024]
  const float* c_attn_w = (const float*)d_in[2];  // [1024,3072]
  const float* c_attn_b = (const float*)d_in[3];  // [3072]
  const float* c_proj_w = (const float*)d_in[4];  // [1024,1024]
  const float* c_proj_b = (const float*)d_in[5];  // [1024]
  const float* gate     = (const float*)d_in[6];  // [16]
  float* out = (float*)d_out;                     // [2,2048,1024]

  // Workspace carving (256B aligned slabs).
  char* ws = (char*)d_ws;
  size_t off = 0;
  auto carve = [&](size_t bytes) -> char* {
    char* p = ws + off;
    off += (bytes + 255) & ~(size_t)255;
    return p;
  };
  float*  qkv  = (float*) carve((size_t)MQKV * N3D * 4);                 // 51.9 MB
  bf16_t* Xp   = (bf16_t*)carve((size_t)MQKV * DIM * 2);                 //  8.7 MB
  bf16_t* Wp   = (bf16_t*)carve((size_t)DIM * N3D * 2);                  //  6.3 MB
  bf16_t* Vp   = (bf16_t*)carve((size_t)32 * 4 * 64 * 512 * 2);          //  8.4 MB
  bf16_t* Vap  = (bf16_t*)carve((size_t)32 * 4 * 2 * 512 * 2);           //  0.3 MB
  float*  attn = (float*) carve((size_t)MROWS * DIM * 4);                // 16.8 MB
  bf16_t* Ap2  = (bf16_t*)carve((size_t)MROWS * DIM * 2);                //  8.4 MB
  bf16_t* Wp2  = (bf16_t*)carve((size_t)DIM * DIM * 2);                  //  2.1 MB

  // 1) Pack operands to bf16 fragment images.
  pack_a_kernel<<<(MROWS * DIM) / 256, 256, 0, stream>>>(
      Xp, hidden, DIM, DIM / 32, MROWS * DIM);
  pack_a_kernel<<<(BATCH * ADA * DIM) / 256, 256, 0, stream>>>(
      Xp + (size_t)(MROWS / 16) * (DIM / 32) * 512, adapter, DIM, DIM / 32,
      BATCH * ADA * DIM);
  pack_b_kernel<<<(DIM * N3D) / 256, 256, 0, stream>>>(
      Wp, c_attn_w, N3D, DIM / 32, DIM * N3D);
  pack_b_kernel<<<(DIM * DIM) / 256, 256, 0, stream>>>(
      Wp2, c_proj_w, DIM, DIM / 32, DIM * DIM);

  // 2) QKV GEMM: [4224,1024] x [1024,3072] + b -> qkv f32.
  gemm_bf16_kernel<<<dim3(MQKV / 64, N3D / 128), 256, 0, stream>>>(
      Xp, Wp, c_attn_b, qkv, MQKV, N3D, DIM);

  // 3) Pack V / adapter-V into per-(b,h) B-fragments.
  pack_v_kernel<<<dim3((4 * 64 * 512) / 256, BATCH * HEADS), 256, 0, stream>>>(
      Vp, qkv, SEQ, 0, SEQ / 32, 4 * 64 * 512, 2 * DIM);
  pack_v_kernel<<<dim3((4 * 2 * 512) / 256, BATCH * HEADS), 256, 0, stream>>>(
      Vap, qkv, ADA, MROWS, ADA / 32, 4 * 2 * 512, 2 * DIM);

  // 4) Flash attention (+ adapter attention + tanh gate).
  attn_kernel<<<dim3(SEQ / 16 / 8, BATCH * HEADS), 256, 0, stream>>>(
      qkv, Vp, Vap, gate, attn);

  // 5) Output projection: pack attn output, then GEMM + bias -> d_out.
  pack_a_kernel<<<(MROWS * DIM) / 256, 256, 0, stream>>>(
      Ap2, attn, DIM, DIM / 32, MROWS * DIM);
  gemm_bf16_kernel<<<dim3(MROWS / 64, DIM / 128), 256, 0, stream>>>(
      Ap2, Wp2, c_proj_b, out, MROWS, DIM, DIM);
}
